// RCNet_Painter_23158463660769
// MI455X (gfx1250) — matrix operand order repaired and checked by reading, not compiled, not run on hardware
//
#include <hip/hip_runtime.h>

// ---------------------------------------------------------------------------
// Complex MLP painter on MI455X (gfx1250), wave32 + v_wmma_f32_16x16x32_f16.
//
// Per wave: 16 points (N dim of WMMA). Computes Y^T = W^T * X^T so the lane
// dimension of B, C and D operands is always "point", making the inter-layer
// activation transpose a pure half-wave swap (ds_swizzle SWAPX16) + cndmask —
// no LDS round-trip for activations. Weights are staged once per block into
// LDS pre-packed in the exact 16-bit A-fragment lane layout.
//
// Round-2 fixes:
//  * laundered LDS offset is a provable multiple of 4 dwords -> ds_load_b128
//    instead of ds_load_2addr_b32 pairs.
//  * final 64->3 layer is now WMMA too (W4^T and -W4i^T staged as zero-padded
//    A fragments) -> removes ~450 VALU + 64 DS loads per tile.
//  * layer 0 uses packed f16 (v_pk_fma) arithmetic.
// ---------------------------------------------------------------------------

typedef unsigned int u32;
typedef __attribute__((ext_vector_type(16))) _Float16 v16h;
typedef __attribute__((ext_vector_type(2)))  _Float16 v2h;
typedef __attribute__((ext_vector_type(8)))  float    v8f;

struct TileArgs {
  const float* zr;
  const float* zi;
  const float* W[5][4];   // [layer][Wr, Wi, br, bi]
  float* out;             // (N, 3) f32
  int ntiles;             // N / 16
};

struct Frag32 { uint4 a, b; };

static __device__ __forceinline__ u32 pk2(float a, float b) {
  union { _Float16 h[2]; u32 u; } x;
  x.h[0] = (_Float16)a; x.h[1] = (_Float16)b;
  return x.u;
}
static __device__ __forceinline__ v2h as_h2(u32 u) {
  return __builtin_bit_cast(v2h, u);
}
// swap lanes l <-> l^16 within wave32 (group-of-32 swizzle, xor=0x10, and=0x1f)
static __device__ __forceinline__ u32 swap16(u32 u) {
  return (u32)__builtin_amdgcn_ds_swizzle((int)u, 0x401f);
}
static __device__ __forceinline__ v16h frag_h(const u32* w) {
  Frag32 f;
  f.a = make_uint4(w[0], w[1], w[2], w[3]);
  f.b = make_uint4(w[4], w[5], w[6], w[7]);
  return __builtin_bit_cast(v16h, f);
}
static __device__ __forceinline__ v8f wmma_f16(v16h a, v16h b, v8f c) {
  // (neg_a, A, neg_b, B, c_mod, C, reuse_a, reuse_b)
  return __builtin_amdgcn_wmma_f32_16x16x32_f16(false, a, false, b, (short)0, c,
                                                false, false);
}

#define THREADS 256
#define WPB 8     // waves per block (wave32)
#define NFRAG 52  // 48 hidden-layer fragments + 4 final-layer fragments

__global__ __launch_bounds__(THREADS)
void rcnet_painter_kernel(TileArgs A) {
  // LDS: weight fragments pre-packed per lane in 16-bit A-matrix layout.
  // Hidden frag id = (layer*2 + mat)*8 + mt*2 + kt   (48 fragments)
  // Final frag id  = 48 + mat*2 + kt                 (mat 0: W4r^T, 1: -W4i^T)
  // Each frag = 256 dwords: [half*128 + lane*4 + j]  (half 0: v0..3, 1: v4..7)
  __shared__ __align__(16) u32   wfrag[NFRAG * 256];       // 52 KB
  __shared__ __align__(16) u32   l0w[128];                 // PWr|PWi|PBr|PBi packed f16
  __shared__ __align__(16) float hbias[3 * 128];           // per layer: br[64] | bi[64]
  __shared__ __align__(16) float fb[16];                   // final bias br4 padded

  const int tid = threadIdx.x;

  // ---------------- one-time weight staging (per block) --------------------
  // A-matrix 16-bit layout used here (contiguous k order, same as B):
  //   lane: M = lane&15, k = 2*v + s + 16*(lane>>4)
  for (int idx = tid; idx < NFRAG * 256; idx += THREADS) {
    int frag = idx >> 8;
    int w    = idx & 255;
    int half = w >> 7;
    int rem  = w & 127;
    int lane = rem >> 2;
    int j    = rem & 3;
    int v    = half * 4 + j;
    int hi   = lane >> 4;
    int kt   = frag & 1;
    int k0   = kt * 32 + (hi << 4) + (v << 1);
    u32 val;
    if (frag < 48) {
      int L   = frag >> 4;         // 0..2 -> params[L+1]
      int mat = (frag >> 3) & 1;   // 0=Wr, 1=Wi
      int mt  = (frag >> 1) & 3;
      int n   = mt * 16 + (lane & 15);
      const float* Wp = A.W[L + 1][mat];  // (64,64) row-major (k_in, n_out)
      val = pk2(Wp[k0 * 64 + n], Wp[(k0 + 1) * 64 + n]);
    } else {
      int g   = frag - 48;
      int mat = g >> 1;            // 0: W4r^T, 1: -W4i^T
      int c   = lane & 15;         // output channel (only 0..2 valid)
      const float* Wp = A.W[4][mat];      // (64,3) row-major
      float s  = mat ? -1.0f : 1.0f;
      float a0 = (c < 3) ? s * Wp[k0 * 3 + c]       : 0.0f;
      float a1 = (c < 3) ? s * Wp[(k0 + 1) * 3 + c] : 0.0f;
      val = pk2(a0, a1);
    }
    wfrag[idx] = val;
  }
  // Layer-0 packed data, indexed exactly as the consumer B fragment:
  //   idx = arr*32 + kt*16 + h*8 + v ; feature f = kt*32 + 16h + 2v
  for (int idx = tid; idx < 128; idx += THREADS) {
    int arr = idx >> 5, sub = idx & 31;
    int kt = sub >> 4, h = (sub >> 3) & 1, v = sub & 7;
    int f = kt * 32 + h * 16 + v * 2;
    const float* s = A.W[0][arr];   // Wr0 / Wi0 / br0 / bi0, each 64 values
    l0w[idx] = pk2(s[f], s[f + 1]);
  }
  // Hidden biases (f32)
  for (int idx = tid; idx < 384; idx += THREADS) {
    int L = idx >> 7, r = idx & 127;
    const float* s = A.W[L + 1][2 + (r >> 6)];
    hbias[idx] = s[r & 63];
  }
  if (tid < 16) fb[tid] = (tid < 3) ? A.W[4][2][tid] : 0.0f;
  __syncthreads();

  const int l  = tid & 31;
  const int h  = l >> 4;       // half-wave id (0/1)
  const int pl = l & 15;       // point within tile
  const int wave   = blockIdx.x * WPB + (tid >> 5);
  const int nwaves = gridDim.x * WPB;

  for (int t = wave; t < A.ntiles; t += nwaves) {
    // Opaque multiple-of-4 dword offset (value 0): keeps LDS loads inside the
    // loop (no 384-VGPR hoist+spill) while preserving 16B-alignment proof so
    // the backend emits ds_load_b128.
    int lz = 0;
    asm volatile("" : "+v"(lz));
    const int lz4 = lz << 2;

    const int p = t * 16 + pl;
    const float zrf = A.zr[p];
    const float zif = A.zi[p];
    const v2h zr2 = { (_Float16)zrf, (_Float16)zrf };
    const v2h zi2 = { (_Float16)zif, (_Float16)zif };

    // ---------------- layer 0 (1 -> 64), direct into B fragments ----------
    // B (K x N=points) layout used here (contiguous k, as A): lane n = l&15, k = 2v + s + 16*(l>>4)
    u32 cxr[2][8], cxi[2][8];
#pragma unroll
    for (int kt = 0; kt < 2; kt++) {
      const int bidx = kt * 16 + h * 8 + lz4;
      u32 wr[8], wi[8], br[8], bi[8];
      *(uint4*)&wr[0] = *(const uint4*)&l0w[bidx];
      *(uint4*)&wr[4] = *(const uint4*)&l0w[bidx + 4];
      *(uint4*)&wi[0] = *(const uint4*)&l0w[32 + bidx];
      *(uint4*)&wi[4] = *(const uint4*)&l0w[32 + bidx + 4];
      *(uint4*)&br[0] = *(const uint4*)&l0w[64 + bidx];
      *(uint4*)&br[4] = *(const uint4*)&l0w[64 + bidx + 4];
      *(uint4*)&bi[0] = *(const uint4*)&l0w[96 + bidx];
      *(uint4*)&bi[4] = *(const uint4*)&l0w[96 + bidx + 4];
#pragma unroll
      for (int v = 0; v < 8; v++) {
        const v2h w2 = as_h2(wr[v]), u2 = as_h2(wi[v]);
        const v2h c2 = as_h2(br[v]), d2 = as_h2(bi[v]);
        const v2h rp = zr2 * w2 - zi2 * u2 + c2;   // v_pk_fma_f16
        const v2h ip = zi2 * w2 + zr2 * u2 + d2;
        cxr[kt][v] = pk2(__sinf((float)rp.x), __sinf((float)rp.y));
        cxi[kt][v] = pk2(__sinf((float)ip.x), __sinf((float)ip.y));
      }
    }

    // ---------------- hidden layers 1..3 (64 -> 64 complex) ----------------
#pragma unroll
    for (int L = 0; L < 3; L++) {
      v8f accR[4], accI[4];
#pragma unroll
      for (int mt = 0; mt < 4; mt++) {
        // C/D layout: VGPR d -> out-feature mt*16 + d + 8h
        const int bb = L * 128 + mt * 16 + h * 8 + lz4;
        union { float f[8]; v8f v; } ur, ui;
        *(float4*)&ur.f[0] = *(const float4*)&hbias[bb];
        *(float4*)&ur.f[4] = *(const float4*)&hbias[bb + 4];
        *(float4*)&ui.f[0] = *(const float4*)&hbias[64 + bb];
        *(float4*)&ui.f[4] = *(const float4*)&hbias[64 + bb + 4];
        accR[mt] = ur.v;
        accI[mt] = ui.v;
      }
#pragma unroll
      for (int kt = 0; kt < 2; kt++) {
        u32 nxi[8];
#pragma unroll
        for (int v = 0; v < 8; v++) nxi[v] = cxi[kt][v] ^ 0x80008000u;
        const v16h Bxr = frag_h(cxr[kt]);
        const v16h Bxi = frag_h(cxi[kt]);
        const v16h Bnx = frag_h(nxi);
#pragma unroll
        for (int mt = 0; mt < 4; mt++) {
          const int fr = ((L * 2 + 0) * 8 + mt * 2 + kt) * 256;
          const int fi = ((L * 2 + 1) * 8 + mt * 2 + kt) * 256;
          const int wb = l * 4 + lz4;
          Frag32 fa, fc;
          fa.a = *(const uint4*)&wfrag[fr + wb];
          fa.b = *(const uint4*)&wfrag[fr + 128 + wb];
          fc.a = *(const uint4*)&wfrag[fi + wb];
          fc.b = *(const uint4*)&wfrag[fi + 128 + wb];
          const v16h Awr = __builtin_bit_cast(v16h, fa);
          const v16h Awi = __builtin_bit_cast(v16h, fc);
          // Yr^T = Wr^T Xr^T - Wi^T Xi^T ;  Yi^T = Wr^T Xi^T + Wi^T Xr^T
          accR[mt] = wmma_f16(Awr, Bxr, accR[mt]);
          accR[mt] = wmma_f16(Awi, Bnx, accR[mt]);
          accI[mt] = wmma_f16(Awr, Bxi, accI[mt]);
          accI[mt] = wmma_f16(Awi, Bxr, accI[mt]);
        }
      }
      // sin activation -> producer-packed f16 (features mt*16 + 8h + 2j, +1),
      // then transpose hand-off (half-wave swap + select) -> next B fragments.
      u32 pr[4][4], pi[4][4], swr[4][4], swi[4][4];
#pragma unroll
      for (int mt = 0; mt < 4; mt++) {
#pragma unroll
        for (int j = 0; j < 4; j++) {
          pr[mt][j] = pk2(__sinf(accR[mt][2 * j]), __sinf(accR[mt][2 * j + 1]));
          pi[mt][j] = pk2(__sinf(accI[mt][2 * j]), __sinf(accI[mt][2 * j + 1]));
          swr[mt][j] = swap16(pr[mt][j]);
          swi[mt][j] = swap16(pi[mt][j]);
        }
      }
#pragma unroll
      for (int kt = 0; kt < 2; kt++)
#pragma unroll
        for (int v = 0; v < 4; v++) {
          cxr[kt][v]     = h ? swr[2 * kt + 1][v] : pr[2 * kt][v];
          cxr[kt][4 + v] = h ? pr[2 * kt + 1][v]  : swr[2 * kt][v];
          cxi[kt][v]     = h ? swi[2 * kt + 1][v] : pi[2 * kt][v];
          cxi[kt][4 + v] = h ? pi[2 * kt + 1][v]  : swi[2 * kt][v];
        }
    }

    // ---------------- final layer (64 -> 3) via WMMA + sigmoid -------------
    // r^T = W4r^T * Xr^T + (-W4i^T) * Xi^T ; channels live in M = d + 8h.
    v8f accF;
    {
      const int bb = h * 8 + lz4;
      union { float f[8]; v8f v; } uf;
      *(float4*)&uf.f[0] = *(const float4*)&fb[bb];
      *(float4*)&uf.f[4] = *(const float4*)&fb[bb + 4];
      accF = uf.v;
    }
#pragma unroll
    for (int kt = 0; kt < 2; kt++) {
      const v16h Bxr = frag_h(cxr[kt]);
      const v16h Bxi = frag_h(cxi[kt]);
      const int wb = l * 4 + lz4;
      const int f0 = (48 + 0 * 2 + kt) * 256;
      const int f1 = (48 + 1 * 2 + kt) * 256;
      Frag32 fa, fc;
      fa.a = *(const uint4*)&wfrag[f0 + wb];
      fa.b = *(const uint4*)&wfrag[f0 + 128 + wb];
      fc.a = *(const uint4*)&wfrag[f1 + wb];
      fc.b = *(const uint4*)&wfrag[f1 + 128 + wb];
      accF = wmma_f16(__builtin_bit_cast(v16h, fa), Bxr, accF);
      accF = wmma_f16(__builtin_bit_cast(v16h, fc), Bxi, accF);
    }
    const float s0 = 1.0f / (1.0f + __expf(-accF[0]));
    const float s1 = 1.0f / (1.0f + __expf(-accF[1]));
    const float s2 = 1.0f / (1.0f + __expf(-accF[2]));
    if (h == 0) {
      float* o = A.out + (size_t)(t * 16 + pl) * 3;
      o[0] = s0; o[1] = s1; o[2] = s2;
    }
  }
}

extern "C" void kernel_launch(void* const* d_in, const int* in_sizes, int n_in,
                              void* d_out, int out_size, void* d_ws, size_t ws_size,
                              hipStream_t stream) {
  (void)n_in; (void)out_size; (void)d_ws; (void)ws_size;
  TileArgs a;
  a.zr = (const float*)d_in[0];
  a.zi = (const float*)d_in[1];
  for (int L = 0; L < 5; L++)
    for (int k = 0; k < 4; k++)
      a.W[L][k] = (const float*)d_in[2 + L * 4 + k];
  a.out = (float*)d_out;
  const int n = in_sizes[0];
  a.ntiles = n / 16;

  const int blocks = 1024;  // 8192 waves, 8 tiles/wave at N = 1M
  rcnet_painter_kernel<<<blocks, THREADS, 0, stream>>>(a);
}
